// WaveletCrossBandBlock_38448547234074
// MI455X (gfx1250) — matrix operand order repaired and checked by reading, not compiled, not run on hardware
//
#include <hip/hip_runtime.h>

typedef __attribute__((ext_vector_type(16))) __bf16 v16bf;
typedef __attribute__((ext_vector_type(8)))  float  v8f;
typedef __attribute__((ext_vector_type(4)))  unsigned int v4u;

union BF16Frag { v16bf v; v4u q[2]; unsigned int u[8]; };

#define WMMA_BF16(A,B,Cc) __builtin_amdgcn_wmma_f32_16x16x32_bf16(false,(A),false,(B),(short)0,(Cc),false,false)

constexpr int kC    = 64;
constexpr int kNH   = 4;
constexpr int kHW   = 784;     // 28*28
constexpr int kDHW  = 21952;   // 28^3
constexpr int kNTok = 343;     // 7^3
constexpr int kBW   = 128;     // 2 batches * 64 windows
constexpr int kB    = 2;

__device__ __forceinline__ unsigned short f2bf(float f) {
    unsigned int u = __float_as_uint(f);
    unsigned int r = u + 0x7FFFu + ((u >> 16) & 1u);
    return (unsigned short)(r >> 16);
}
__device__ __forceinline__ unsigned int pack2(unsigned short lo, unsigned short hi) {
    return (unsigned int)lo | ((unsigned int)hi << 16);
}
__device__ __forceinline__ unsigned int packf(float lo, float hi) {
    return pack2(f2bf(lo), f2bf(hi));
}
__device__ __forceinline__ int regionOf(int x) {   // shifted-window mask region, axis 28, ws 7, ss 3
    return (x < 21) ? 0 : ((x < 25) ? 1 : 2);
}
__device__ __forceinline__ v4u ldsv4(const unsigned int* p) { return *(const v4u*)p; }

// ---------------------------------------------------------------------------
// 1) Per-band per-channel statistics: mean, std, abs-mean, energy
// ---------------------------------------------------------------------------
__global__ __launch_bounds__(256) void stats_kernel(
    const float* __restrict__ p0, const float* __restrict__ p1, const float* __restrict__ p2,
    const float* __restrict__ p3, const float* __restrict__ p4, const float* __restrict__ p5,
    const float* __restrict__ p6, float* __restrict__ raw)
{
    const int bx   = blockIdx.x;
    const int band = bx / (kB * kC);
    const int rem  = bx % (kB * kC);
    const int b    = rem / kC, c = rem % kC;
    const float* bands[7] = {p0, p1, p2, p3, p4, p5, p6};
    const float* src = bands[band] + ((size_t)b * kC + c) * kDHW;

    float s = 0.f, sa = 0.f, s2 = 0.f;
    for (int i = threadIdx.x; i < kDHW; i += 256) {
        float x = src[i];
        s += x; sa += fabsf(x); s2 += x * x;
    }
    __shared__ float rs[256], ra[256], r2[256];
    const int t = threadIdx.x;
    rs[t] = s; ra[t] = sa; r2[t] = s2;
    __syncthreads();
    for (int off = 128; off > 0; off >>= 1) {
        if (t < off) { rs[t] += rs[t+off]; ra[t] += ra[t+off]; r2[t] += r2[t+off]; }
        __syncthreads();
    }
    if (t == 0) {
        const float inv = 1.f / (float)kDHW;
        float mean = rs[0] * inv;
        float en   = r2[0] * inv;
        float var  = en - mean * mean;
        float stdv = sqrtf(var + 1e-6f);
        float am   = ra[0] * inv;
        float* dst = raw + ((size_t)b * 7 + band) * 256;
        dst[c] = mean; dst[64 + c] = stdv; dst[128 + c] = am; dst[192 + c] = en;
    }
}

// ---------------------------------------------------------------------------
// 2) Band graph fusion (7 nodes, C=64) -> gates g[b][band][c]
// ---------------------------------------------------------------------------
__global__ __launch_bounds__(256) void graph_kernel(
    const float* __restrict__ raw_g,
    const float* __restrict__ dp_w,  const float* __restrict__ dp_b,
    const float* __restrict__ n1_g,  const float* __restrict__ n1_b,
    const float* __restrict__ qkv_w, const float* __restrict__ qkv_b,
    const float* __restrict__ pr_w,  const float* __restrict__ pr_b,
    const float* __restrict__ fn_g,  const float* __restrict__ fn_b,
    const float* __restrict__ f1_w,  const float* __restrict__ f1_b,
    const float* __restrict__ f2_w,  const float* __restrict__ f2_b,
    const float* __restrict__ gt_w,  const float* __restrict__ gt_b,
    float* __restrict__ g_out)
{
    const int b = blockIdx.x;
    const int t = threadIdx.x;
    __shared__ float raw_s[7 * 256];
    __shared__ float sim[7][7];
    __shared__ float maskm[7][7];
    __shared__ float nodes0[7][64];
    __shared__ float nodes[7][64];
    __shared__ float qkv[7][192];
    __shared__ float aout[7][64];
    __shared__ float h1[7][256];

    for (int i = t; i < 7 * 256; i += 256) raw_s[i] = raw_g[(size_t)b * 7 * 256 + i];
    __syncthreads();

    if (t < 49) {
        int i = t / 7, j = t % 7;
        float ni = 0.f, nj = 0.f, d = 0.f;
        for (int k = 0; k < 256; k++) {
            float a = raw_s[i*256 + k], c2 = raw_s[j*256 + k];
            ni += a * a; nj += c2 * c2; d += a * c2;
        }
        ni = fmaxf(sqrtf(ni), 1e-6f); nj = fmaxf(sqrtf(nj), 1e-6f);
        sim[i][j] = (i == j) ? -1e30f : d / (ni * nj);
    }
    __syncthreads();

    if (t < 7) {
        float best0 = -1e38f, best1 = -1e38f, best2 = -1e38f;
        int bi0 = -1, bi1 = -1, bi2 = -1;
        for (int j = 0; j < 7; j++) {
            float v = sim[t][j];
            if (v > best0)      { best2=best1; bi2=bi1; best1=best0; bi1=bi0; best0=v; bi0=j; }
            else if (v > best1) { best2=best1; bi2=bi1; best1=v; bi1=j; }
            else if (v > best2) { best2=v; bi2=j; }
        }
        for (int j = 0; j < 7; j++) maskm[t][j] = -1e30f;
        if (bi0 >= 0) maskm[t][bi0] = 0.f;
        if (bi1 >= 0) maskm[t][bi1] = 0.f;
        if (bi2 >= 0) maskm[t][bi2] = 0.f;
    }
    __syncthreads();

    for (int o = t; o < 7 * 64; o += 256) {
        int i = o / 64, c = o % 64;
        float acc = dp_b[c];
        for (int k = 0; k < 256; k++) acc += raw_s[i*256 + k] * dp_w[k*64 + c];
        nodes0[i][c] = acc;
    }
    __syncthreads();

    if (t < 7) {
        float mu = 0.f;
        for (int c = 0; c < 64; c++) mu += nodes0[t][c];
        mu *= (1.f/64.f);
        float v2 = 0.f;
        for (int c = 0; c < 64; c++) { float d = nodes0[t][c]-mu; v2 += d*d; }
        float inv = rsqrtf(v2*(1.f/64.f) + 1e-5f);
        for (int c = 0; c < 64; c++) nodes[t][c] = (nodes0[t][c]-mu)*inv*n1_g[c] + n1_b[c];
    }
    __syncthreads();

    for (int o = t; o < 7 * 192; o += 256) {
        int i = o / 192, c = o % 192;
        float acc = qkv_b[c];
        for (int k = 0; k < 64; k++) acc += nodes[i][k] * qkv_w[k*192 + c];
        qkv[i][c] = acc;
    }
    __syncthreads();

    if (t < 28) {
        int h = t / 7, i = t % 7;
        float s[7]; float mx = -1e38f;
        for (int j = 0; j < 7; j++) {
            float d = 0.f;
            for (int dd = 0; dd < 16; dd++) d += qkv[i][h*16+dd] * qkv[j][64 + h*16 + dd];
            s[j] = d * 0.25f + maskm[i][j];
            mx = fmaxf(mx, s[j]);
        }
        float l = 0.f;
        for (int j = 0; j < 7; j++) { s[j] = __expf(s[j]-mx); l += s[j]; }
        float invl = 1.f / l;
        for (int dd = 0; dd < 16; dd++) {
            float o2 = 0.f;
            for (int j = 0; j < 7; j++) o2 += s[j] * qkv[j][128 + h*16 + dd];
            aout[i][h*16+dd] = o2 * invl;
        }
    }
    __syncthreads();

    for (int o = t; o < 7 * 64; o += 256) {
        int i = o / 64, c = o % 64;
        float acc = pr_b[c];
        for (int k = 0; k < 64; k++) acc += aout[i][k] * pr_w[k*64 + c];
        nodes[i][c] = nodes0[i][c] + acc;
    }
    __syncthreads();

    if (t < 7) {
        float mu = 0.f;
        for (int c = 0; c < 64; c++) mu += nodes[t][c];
        mu *= (1.f/64.f);
        float v2 = 0.f;
        for (int c = 0; c < 64; c++) { float d = nodes[t][c]-mu; v2 += d*d; }
        float inv = rsqrtf(v2*(1.f/64.f) + 1e-5f);
        for (int c = 0; c < 64; c++) aout[t][c] = (nodes[t][c]-mu)*inv*fn_g[c] + fn_b[c];
    }
    __syncthreads();

    for (int o = t; o < 7 * 256; o += 256) {
        int i = o / 256, c = o % 256;
        float acc = f1_b[c];
        for (int k = 0; k < 64; k++) acc += aout[i][k] * f1_w[k*256 + c];
        h1[i][c] = 0.5f * acc * (1.f + erff(acc * 0.70710678118f));
    }
    __syncthreads();

    for (int o = t; o < 7 * 64; o += 256) {
        int i = o / 64, c = o % 64;
        float acc = f2_b[c];
        for (int k = 0; k < 256; k++) acc += h1[i][k] * f2_w[k*64 + c];
        nodes[i][c] += acc;
    }
    __syncthreads();

    for (int o = t; o < 7 * 64; o += 256) {
        int i = o / 64, c = o % 64;
        float acc = gt_b[c];
        for (int k = 0; k < 64; k++) acc += nodes[i][k] * gt_w[k*64 + c];
        g_out[((size_t)b*7 + i)*64 + c] = 1.f / (1.f + __expf(-acc));
    }
}

// ---------------------------------------------------------------------------
// 3) Fusion GEMM: Ff = (details * gate) @ fuse_w + fuse_b
// ---------------------------------------------------------------------------
__global__ __launch_bounds__(256) void fuse_kernel(
    const float* __restrict__ p0, const float* __restrict__ p1, const float* __restrict__ p2,
    const float* __restrict__ p3, const float* __restrict__ p4, const float* __restrict__ p5,
    const float* __restrict__ p6,
    const float* __restrict__ g, const float* __restrict__ fuse_w, const float* __restrict__ fuse_b,
    float* __restrict__ Ff)
{
    const int blkPerB = (kDHW + 127) / 128;           // 172
    const int b  = blockIdx.x / blkPerB;
    const int vb = (blockIdx.x % blkPerB) * 128;
    const float* bands[7] = {p0, p1, p2, p3, p4, p5, p6};
    const bool fullBlk = (vb + 128 <= kDHW);

    __shared__ __align__(16) unsigned int a_pk[128][16];  // [m][K/2]  8KB
    __shared__ __align__(16) unsigned int w_pk[64][16];   // [n][K/2]  4KB

    const int t = threadIdx.x;
    const int wave = t >> 5, lane = t & 31, lh = lane >> 4, nl = lane & 15;

    v8f acc[4];
    #pragma unroll
    for (int i = 0; i < 4; i++) acc[i] = (v8f){0.f,0.f,0.f,0.f,0.f,0.f,0.f,0.f};

    for (int k0 = 0; k0 < 448; k0 += 32) {
        __syncthreads();
        const int band = k0 >> 6;                      // 32-chunks never cross a band
        const float* bp = bands[band] + ((size_t)b * kC) * kDHW;
        const float* gp = g + ((size_t)b * 7 + band) * 64;
        if (fullBlk) {
            for (int q = t; q < 128 * 16; q += 256) {
                int kp = q >> 7, m = q & 127;
                int cin = (k0 & 63) + kp * 2;
                int vox = vb + m;
                a_pk[m][kp] = packf(bp[(size_t)cin * kDHW + vox] * gp[cin],
                                    bp[(size_t)(cin+1) * kDHW + vox] * gp[cin+1]);
            }
        } else {
            for (int q = t; q < 128 * 16; q += 256) {
                int kp = q >> 7, m = q & 127;
                int cin = (k0 & 63) + kp * 2;
                int vox = vb + m;
                unsigned int pk = 0u;
                if (vox < kDHW)
                    pk = packf(bp[(size_t)cin * kDHW + vox] * gp[cin],
                               bp[(size_t)(cin+1) * kDHW + vox] * gp[cin+1]);
                a_pk[m][kp] = pk;
            }
        }
        for (int q = t; q < 64 * 16; q += 256) {       // W tile, n fastest
            int kp = q >> 6, n = q & 63;
            w_pk[n][kp] = packf(fuse_w[(k0 + 2*kp) * 64 + n],
                                fuse_w[(k0 + 2*kp + 1) * 64 + n]);
        }
        __syncthreads();

        const int mrow = wave * 16 + nl;
        BF16Frag a;
        a.q[0] = ldsv4(&a_pk[mrow][lh*8]);
        a.q[1] = ldsv4(&a_pk[mrow][lh*8 + 4]);
        #pragma unroll
        for (int nt = 0; nt < 4; nt++) {
            BF16Frag bf;
            const unsigned int* wp = &w_pk[nt*16 + nl][lh*8];
            bf.q[0] = ldsv4(wp);
            bf.q[1] = ldsv4(wp + 4);
            acc[nt] = WMMA_BF16(a.v, bf.v, acc[nt]);
        }
    }
    if (fullBlk) {
        #pragma unroll
        for (int nt = 0; nt < 4; nt++) {
            const int c = nt * 16 + nl;
            float* dst = Ff + ((size_t)b * kC + c) * kDHW + vb + wave * 16 + lh * 8;
            #pragma unroll
            for (int r = 0; r < 8; r++) dst[r] = acc[nt][r] + fuse_b[c];
        }
    } else {
        #pragma unroll
        for (int nt = 0; nt < 4; nt++) {
            const int c = nt * 16 + nl;
            #pragma unroll
            for (int r = 0; r < 8; r++) {
                int vox = vb + wave * 16 + lh * 8 + r;
                if (vox < kDHW)
                    Ff[((size_t)b * kC + c) * kDHW + vox] = acc[nt][r] + fuse_b[c];
            }
        }
    }
}

// ---------------------------------------------------------------------------
// 4) CSWA stage 1: fused LN + roll(-3) + window partition + Q/KV projection
// ---------------------------------------------------------------------------
__global__ __launch_bounds__(256) void qkv_kernel(
    const float* __restrict__ xq, const float* __restrict__ xkv,
    const float* __restrict__ nqg, const float* __restrict__ nqb,
    const float* __restrict__ nkg, const float* __restrict__ nkb,
    const float* __restrict__ qw,  const float* __restrict__ qb,
    const float* __restrict__ kvw, const float* __restrict__ kvb,
    unsigned short* __restrict__ Qb, unsigned short* __restrict__ Kb,
    unsigned short* __restrict__ Vb)
{
    const int widx = blockIdx.x;
    const int b = widx >> 6, wl = widx & 63;
    const int wd = wl >> 4, wh = (wl >> 2) & 3, ww = wl & 3;

    __shared__ __align__(16) unsigned int x_pk[256][32];  // [tok][K/2]  32KB
    __shared__ __align__(16) unsigned int w_pk[192][32];  // [n][K/2]    24KB
    __shared__ float bia[192];

    const int t = threadIdx.x;
    for (int i = t; i < 64 * 32; i += 256) {              // qw -> cols 0..63
        int kp = i >> 6, n = i & 63;
        w_pk[n][kp] = packf(qw[2*kp*64 + n], qw[(2*kp+1)*64 + n]);
    }
    for (int i = t; i < 128 * 32; i += 256) {             // kvw -> cols 64..191
        int kp = i >> 7, n = i & 127;
        w_pk[64 + n][kp] = packf(kvw[2*kp*128 + n], kvw[(2*kp+1)*128 + n]);
    }
    for (int i = t; i < 192; i += 256) bia[i] = (i < 64) ? qb[i] : kvb[i - 64];

    const int wave = t >> 5, lane = t & 31, lh = lane >> 4, nl = lane & 15;

    for (int tile = 0; tile < 2; ++tile) {
        const int tok0 = tile * 256;
        for (int sel = 0; sel < 2; ++sel) {        // 0: q source, 1: kv source
            __syncthreads();
            {   // LayerNorm: one token per thread, packed bf16 row to LDS
                const int tok = tok0 + t;
                const float* src = sel ? xkv : xq;
                const float* gg  = sel ? nkg : nqg;
                const float* bbv = sel ? nkb : nqb;
                if (tok < kNTok) {
                    const int td = tok / 49, th = (tok / 7) % 7, tw = tok % 7;
                    const int sd = (wd*7 + td + 3) % 28;
                    const int sh = (wh*7 + th + 3) % 28;
                    const int sw = (ww*7 + tw + 3) % 28;
                    const size_t base = (size_t)b * kC * kDHW + (size_t)sd * kHW + (size_t)sh * 28 + sw;
                    float mu = 0.f, s2 = 0.f;
                    for (int c = 0; c < 64; c++) { float x = src[base + (size_t)c * kDHW]; mu += x; s2 += x * x; }
                    mu *= (1.f/64.f);
                    float var = s2 * (1.f/64.f) - mu * mu;
                    float inv = rsqrtf(var + 1e-5f);
                    for (int c = 0; c < 64; c += 2) {
                        float x0 = src[base + (size_t)c * kDHW];
                        float x1 = src[base + (size_t)(c+1) * kDHW];
                        x_pk[t][c >> 1] = packf((x0 - mu) * inv * gg[c]   + bbv[c],
                                                (x1 - mu) * inv * gg[c+1] + bbv[c+1]);
                    }
                } else {
                    for (int kp = 0; kp < 32; kp++) x_pk[t][kp] = 0u;
                }
            }
            __syncthreads();

            const int npairs = sel ? 128 : 64;     // (m-subtile, n-tile) work items
            for (int pr = wave; pr < npairs; pr += 8) {
                int msub, nt;
                if (sel) { msub = pr >> 3; nt = pr & 7; }
                else     { msub = pr >> 2; nt = pr & 3; }
                const int ncol0 = (sel ? 64 : 0) + nt * 16;
                const int mrow  = msub * 16 + nl;
                const int nc    = ncol0 + nl;
                v8f acc = (v8f){0.f,0.f,0.f,0.f,0.f,0.f,0.f,0.f};
                #pragma unroll
                for (int k0 = 0; k0 < 64; k0 += 32) {
                    BF16Frag a, bf;
                    a.q[0] = ldsv4(&x_pk[mrow][(k0>>1) + lh*8]);
                    a.q[1] = ldsv4(&x_pk[mrow][(k0>>1) + lh*8 + 4]);
                    bf.q[0] = ldsv4(&w_pk[nc][(k0>>1) + lh*8]);
                    bf.q[1] = ldsv4(&w_pk[nc][(k0>>1) + lh*8 + 4]);
                    acc = WMMA_BF16(a.v, bf.v, acc);
                }
                const int c = nc;
                const int tbase = tok0 + msub * 16;
                const bool full = (tbase + 16) <= kNTok;
                unsigned short* dst;
                float scale, shift;
                if (sel == 0) {
                    int head = c >> 4, dd = c & 15;
                    dst = Qb + (((size_t)widx*4 + head)*kNTok + tbase)*16 + dd;
                    scale = 0.25f; shift = bia[c] * 0.25f;      // (acc+bias)*dh^-0.5
                } else {
                    int c2 = c - 64, pv = c2 >> 6, head = (c2 >> 4) & 3, dd = c2 & 15;
                    dst = (pv ? Vb : Kb) + (((size_t)widx*4 + head)*kNTok + tbase)*16 + dd;
                    scale = 1.f; shift = bia[c];
                }
                if (full) {
                    #pragma unroll
                    for (int r = 0; r < 8; ++r)
                        dst[(size_t)(lh*8 + r)*16] = f2bf(acc[r] * scale + shift);
                } else {
                    #pragma unroll
                    for (int r = 0; r < 8; ++r) {
                        int m = lh * 8 + r;
                        if (tbase + m < kNTok)
                            dst[(size_t)m*16] = f2bf(acc[r] * scale + shift);
                    }
                }
            }
        }
    }
}

// ---------------------------------------------------------------------------
// 5) CSWA stage 2: flash attention per (window, head). rpb + shift-mask fused.
// K tile staged with CDNA5 async global->LDS DMA (ASYNCcnt).
// ---------------------------------------------------------------------------
__global__ __launch_bounds__(256) void attn_kernel(
    const unsigned short* __restrict__ Qb, const unsigned short* __restrict__ Kb,
    const unsigned short* __restrict__ Vb, const float* __restrict__ rpb,
    unsigned short* __restrict__ OW)
{
    const int blk  = blockIdx.x;
    const int widx = blk >> 2, head = blk & 3;
    const int wl = widx & 63;
    const int wd = wl >> 4, wh = (wl >> 2) & 3, ww = wl & 3;

    __shared__ __align__(16) unsigned int k_pk[352][8];    // [key][dh/2]   11264B
    __shared__ __align__(16) unsigned int v_pk[16][176];   // [dh][key/2]   11264B
    __shared__ __align__(16) unsigned int p_pk[8][16][16]; // [wave][m][key/2] 8KB
    __shared__ float rpb_s[2197];
    __shared__ float S_s[8][16][32];
    __shared__ float alpha_s[8][16];
    __shared__ float l_s[8][16];
    __shared__ float m_s[8][16];
    __shared__ unsigned int tinfo[352];  // d | h<<4 | w<<8 | region<<12 | valid<<31

    const int t = threadIdx.x;
    const unsigned short* Kg = Kb + (((size_t)widx*4 + head) * kNTok) * 16;
    const unsigned short* Vg = Vb + (((size_t)widx*4 + head) * kNTok) * 16;

    // --- K rows: async global->LDS b128 copies (343*32B = 686 x 16B).
    //     Padding rows 343..351 stay stale: every S element with key>=343 is
    //     overwritten with -1e30 before softmax, so the garbage never escapes.
    {
        const unsigned int ldsBase = (unsigned int)(unsigned long long)(&k_pk[0][0]);
        for (int i = t; i < 686; i += 256) {
            unsigned int ldsOff = ldsBase + (unsigned int)i * 16u;
            unsigned int gOff   = (unsigned int)i * 16u;
            asm volatile("global_load_async_to_lds_b128 %0, %1, %2"
                         :: "v"(ldsOff), "v"(gOff), "s"(Kg) : "memory");
        }
    }
    for (int i = t; i < 16 * 176; i += 256) {              // V: transpose-pack over keys
        int kp = i >> 4, n = i & 15;
        unsigned short lo = (2*kp     < kNTok) ? Vg[(2*kp)  *16 + n] : (unsigned short)0;
        unsigned short hi = (2*kp + 1 < kNTok) ? Vg[(2*kp+1)*16 + n] : (unsigned short)0;
        v_pk[n][kp] = pack2(lo, hi);
    }
    for (int i = t; i < 2197; i += 256) rpb_s[i] = rpb[i*4 + head];
    for (int i = t; i < 352; i += 256) {
        int dc = i / 49, hc = (i / 7) % 7, wc = i % 7;
        unsigned int reg = (unsigned int)(regionOf(wd*7 + dc)*9 + regionOf(wh*7 + hc)*3 + regionOf(ww*7 + wc));
        tinfo[i] = (i < kNTok)
                 ? ((unsigned int)dc | ((unsigned int)hc << 4) | ((unsigned int)wc << 8) | (reg << 12) | 0x80000000u)
                 : 0u;
    }
    asm volatile("s_wait_asynccnt 0" ::: "memory");
    __syncthreads();

    const int wave = t >> 5, lane = t & 31, lh = lane >> 4, nl = lane & 15;
    const v4u zero4 = (v4u){0u,0u,0u,0u};

    for (int mt = wave; mt < 22; mt += 8) {
        const int q0 = mt * 16;
        // Q fragment (K = dh = 16, upper half zero): one b128 global load
        BF16Frag qa;
        {
            const int row = q0 + nl;
            if (row < kNTok) {
                const v4u* qrp = (const v4u*)(Qb + (((size_t)widx*4 + head)*kNTok + row)*16);
                qa.q[0] = qrp[lh];
            } else {
                qa.q[0] = zero4;
            }
            qa.q[1] = zero4;
        }
        // row-side token info for this tile (8 rows per lane-half)
        unsigned int rinfo[8];
        #pragma unroll
        for (int r = 0; r < 8; r++) rinfo[r] = tinfo[q0 + lh*8 + r];

        if (lane < 16) { m_s[wave][lane] = -1e30f; l_s[wave][lane] = 0.f; }
        asm volatile("s_wait_dscnt 0" ::: "memory");
        v8f oacc = (v8f){0.f,0.f,0.f,0.f,0.f,0.f,0.f,0.f};

        for (int kc = 0; kc < 352; kc += 32) {
            // ---- S = Q K^T for 32 keys (two 16-wide WMMAs), bias fused on store
            #pragma unroll
            for (int half = 0; half < 2; half++) {
                const int key = kc + half*16 + nl;
                BF16Frag bf;
                bf.q[0] = ldsv4(&k_pk[key][lh*4]);
                bf.q[1] = zero4;
                v8f sacc = (v8f){0.f,0.f,0.f,0.f,0.f,0.f,0.f,0.f};
                sacc = WMMA_BF16(qa.v, bf.v, sacc);
                const unsigned int kin = tinfo[key];
                #pragma unroll
                for (int r = 0; r < 8; r++) {
                    const unsigned int ri = rinfo[r];
                    float sv;
                    if (ri & kin & 0x80000000u) {
                        int dd = (int)(ri & 15u)        - (int)(kin & 15u)        + 6;
                        int dh = (int)((ri >> 4) & 15u) - (int)((kin >> 4) & 15u) + 6;
                        int dw = (int)((ri >> 8) & 15u) - (int)((kin >> 8) & 15u) + 6;
                        int idx = (dd * 13 + dh) * 13 + dw;
                        sv = sacc[r] + rpb_s[idx];
                        if (((ri >> 12) & 31u) != ((kin >> 12) & 31u)) sv -= 100.f;
                    } else {
                        sv = -1e30f;
                    }
                    S_s[wave][lh*8 + r][half*16 + nl] = sv;
                }
            }
            asm volatile("s_wait_dscnt 0" ::: "memory");

            // ---- online softmax (lanes 0..15 each own one row), write packed P
            if (lane < 16) {
                const int m = lane;
                float cm = -1e30f;
                #pragma unroll
                for (int j = 0; j < 32; j++) cm = fmaxf(cm, S_s[wave][m][j]);
                float old_m = m_s[wave][m];
                float nm = fmaxf(old_m, cm);
                float al = __expf(old_m - nm);
                float ps = 0.f;
                #pragma unroll
                for (int j = 0; j < 32; j += 2) {
                    float pe0 = __expf(S_s[wave][m][j]     - nm);
                    float pe1 = __expf(S_s[wave][m][j + 1] - nm);
                    p_pk[wave][m][j >> 1] = packf(pe0, pe1);
                    ps += pe0 + pe1;
                }
                l_s[wave][m] = l_s[wave][m] * al + ps;
                m_s[wave][m] = nm;
                alpha_s[wave][m] = al;
            }
            asm volatile("s_wait_dscnt 0" ::: "memory");

            // ---- rescale running O, then O += P @ V (full K=32 WMMA)
            #pragma unroll
            for (int r = 0; r < 8; r++) oacc[r] *= alpha_s[wave][lh*8 + r];

            BF16Frag pa, vb2;
            pa.q[0] = ldsv4(&p_pk[wave][nl][lh*8]);
            pa.q[1] = ldsv4(&p_pk[wave][nl][lh*8 + 4]);
            const unsigned int* vr = &v_pk[nl][(kc >> 1) + lh*8];
            vb2.q[0] = ldsv4(vr);
            vb2.q[1] = ldsv4(vr + 4);
            oacc = WMMA_BF16(pa.v, vb2.v, oacc);
        }

        // ---- normalize and write out_win (bf16) [win][tok][head*16+dh]
        unsigned short* obase = OW + ((size_t)widx * kNTok + q0 + lh*8) * 64 + head*16 + nl;
        if (q0 + 16 <= kNTok) {
            #pragma unroll
            for (int r = 0; r < 8; r++) {
                float ov = oacc[r] / fmaxf(l_s[wave][lh*8 + r], 1e-30f);
                obase[(size_t)r * 64] = f2bf(ov);
            }
        } else {
            #pragma unroll
            for (int r = 0; r < 8; r++) {
                int row = q0 + lh*8 + r;
                if (row < kNTok) {
                    float ov = oacc[r] / fmaxf(l_s[wave][lh*8 + r], 1e-30f);
                    obase[(size_t)r * 64] = f2bf(ov);
                }
            }
        }
    }
}

// ---------------------------------------------------------------------------
// 6) CSWA stage 3: out-proj GEMM + window reverse + roll(+3) + residual
// ---------------------------------------------------------------------------
__global__ __launch_bounds__(256) void proj_kernel(
    const unsigned short* __restrict__ OW, const float* __restrict__ pw,
    const float* __restrict__ pb, const float* __restrict__ xq,
    float* __restrict__ out)
{
    const int widx = blockIdx.x;
    const int b = widx >> 6, wl = widx & 63;
    const int wd = wl >> 4, wh = (wl >> 2) & 3, ww = wl & 3;

    __shared__ __align__(16) unsigned int w_pk[64][32];   // [n][K/2]  8KB
    __shared__ float pb_s[64];
    const int t = threadIdx.x;
    for (int i = t; i < 64 * 32; i += 256) {
        int kp = i >> 6, n = i & 63;
        w_pk[n][kp] = packf(pw[2*kp*64 + n], pw[(2*kp+1)*64 + n]);
    }
    for (int i = t; i < 64; i += 256) pb_s[i] = pb[i];
    __syncthreads();

    const int wave = t >> 5, lane = t & 31, lh = lane >> 4, nl = lane & 15;
    const v4u zero4 = (v4u){0u,0u,0u,0u};

    for (int mt = wave; mt < 22; mt += 8) {
        const int tok0 = mt * 16;
        v8f acc[4];
        #pragma unroll
        for (int i = 0; i < 4; i++) acc[i] = (v8f){0.f,0.f,0.f,0.f,0.f,0.f,0.f,0.f};

        #pragma unroll
        for (int k0 = 0; k0 < 64; k0 += 32) {
            BF16Frag a;
            const int row = tok0 + nl;
            if (row < kNTok) {
                const v4u* ar = (const v4u*)(OW + ((size_t)widx * kNTok + row) * 64);
                a.q[0] = ar[(k0 >> 3) + 2*lh];
                a.q[1] = ar[(k0 >> 3) + 2*lh + 1];
            } else {
                a.q[0] = zero4; a.q[1] = zero4;
            }
            #pragma unroll
            for (int nt = 0; nt < 4; nt++) {
                BF16Frag bf;
                const unsigned int* wp = &w_pk[nt*16 + nl][(k0 >> 1) + lh*8];
                bf.q[0] = ldsv4(wp);
                bf.q[1] = ldsv4(wp + 4);
                acc[nt] = WMMA_BF16(a.v, bf.v, acc[nt]);
            }
        }
        const bool full = (tok0 + 16) <= kNTok;
        #pragma unroll
        for (int nt = 0; nt < 4; nt++) {
            const int c = nt * 16 + nl;
            #pragma unroll
            for (int r = 0; r < 8; r++) {
                int m = lh*8 + r;
                int tok = tok0 + m;
                if (!full && tok >= kNTok) continue;
                int td = tok / 49, th = (tok / 7) % 7, tw = tok % 7;
                int sd = (wd*7 + td + 3) % 28;
                int sh = (wh*7 + th + 3) % 28;
                int sw = (ww*7 + tw + 3) % 28;
                size_t gi = ((size_t)b * kC + c) * kDHW + (size_t)sd * kHW + (size_t)sh * 28 + sw;
                out[gi] = acc[nt][r] + pb_s[c] + xq[gi];
            }
        }
    }
}

// ---------------------------------------------------------------------------
// Host launcher
// ---------------------------------------------------------------------------
extern "C" void kernel_launch(void* const* d_in, const int* in_sizes, int n_in,
                              void* d_out, int out_size, void* d_ws, size_t ws_size,
                              hipStream_t stream) {
    (void)in_sizes; (void)n_in; (void)out_size; (void)ws_size;

    const float* LLL = (const float*)d_in[0];
    const float* bands[7];
    for (int i = 0; i < 7; i++) bands[i] = (const float*)d_in[1 + i];
    const float* dp_w   = (const float*)d_in[8];
    const float* dp_b   = (const float*)d_in[9];
    const float* n1_g   = (const float*)d_in[10];
    const float* n1_b   = (const float*)d_in[11];
    const float* qkv_w  = (const float*)d_in[12];
    const float* qkv_b  = (const float*)d_in[13];
    const float* pr_w   = (const float*)d_in[14];
    const float* pr_b   = (const float*)d_in[15];
    const float* fn_g   = (const float*)d_in[16];
    const float* fn_b   = (const float*)d_in[17];
    const float* f1_w   = (const float*)d_in[18];
    const float* f1_b   = (const float*)d_in[19];
    const float* f2_w   = (const float*)d_in[20];
    const float* f2_b   = (const float*)d_in[21];
    const float* gt_w   = (const float*)d_in[22];
    const float* gt_b   = (const float*)d_in[23];
    const float* fuse_w = (const float*)d_in[24];
    const float* fuse_b = (const float*)d_in[25];
    const float* a_nq_g = (const float*)d_in[26];
    const float* a_nq_b = (const float*)d_in[27];
    const float* a_nkv_g= (const float*)d_in[28];
    const float* a_nkv_b= (const float*)d_in[29];
    const float* a_rpb  = (const float*)d_in[30];
    const float* a_q_w  = (const float*)d_in[31];
    const float* a_q_b  = (const float*)d_in[32];
    const float* a_kv_w = (const float*)d_in[33];
    const float* a_kv_b = (const float*)d_in[34];
    const float* a_pr_w = (const float*)d_in[35];
    const float* a_pr_b = (const float*)d_in[36];

    // workspace layout (~34 MB)
    char* ws = (char*)d_ws;
    float* raw  = (float*)ws;                               // B*7*256 f32
    float* gbuf = (float*)(ws + 16384);                     // B*7*64  f32
    float* Ff   = (float*)(ws + 32768);                     // B*C*DHW f32
    size_t o = 32768 + (size_t)kB * kC * kDHW * 4;
    const size_t qkvBytes = (size_t)kBW * kNTok * 64 * 2;
    unsigned short* Qb = (unsigned short*)(ws + o); o += qkvBytes;
    unsigned short* Kb = (unsigned short*)(ws + o); o += qkvBytes;
    unsigned short* Vb = (unsigned short*)(ws + o); o += qkvBytes;
    unsigned short* OW = (unsigned short*)(ws + o);

    // --- band graph fusion ---
    stats_kernel<<<7 * kB * kC, 256, 0, stream>>>(
        bands[0], bands[1], bands[2], bands[3], bands[4], bands[5], bands[6], raw);
    graph_kernel<<<kB, 256, 0, stream>>>(
        raw, dp_w, dp_b, n1_g, n1_b, qkv_w, qkv_b, pr_w, pr_b,
        fn_g, fn_b, f1_w, f1_b, f2_w, f2_b, gt_w, gt_b, gbuf);
    fuse_kernel<<<kB * ((kDHW + 127) / 128), 256, 0, stream>>>(
        bands[0], bands[1], bands[2], bands[3], bands[4], bands[5], bands[6],
        gbuf, fuse_w, fuse_b, Ff);

    float* out = (float*)d_out;
    const size_t slot = (size_t)kB * kC * kDHW;

    auto cswa = [&](const float* xq, const float* xkv, int cfg, float* op) {
        qkv_kernel<<<kBW, 256, 0, stream>>>(
            xq, xkv,
            a_nq_g + cfg * 64, a_nq_b + cfg * 64, a_nkv_g + cfg * 64, a_nkv_b + cfg * 64,
            a_q_w + cfg * 4096, a_q_b + cfg * 64, a_kv_w + cfg * 8192, a_kv_b + cfg * 128,
            Qb, Kb, Vb);
        attn_kernel<<<kBW * kNH, 256, 0, stream>>>(
            Qb, Kb, Vb, a_rpb + (size_t)cfg * 2197 * 4, OW);
        proj_kernel<<<kBW, 256, 0, stream>>>(
            OW, a_pr_w + cfg * 4096, a_pr_b + cfg * 64, xq, op);
    };

    cswa(LLL,      Ff,  0, out);               // LLL_new
    cswa(bands[0], out, 1, out + 1 * slot);    // LLH  vs LLL_new
    cswa(bands[1], out, 1, out + 2 * slot);    // LHL
    cswa(bands[2], out, 1, out + 3 * slot);    // HLL
    cswa(bands[3], out, 2, out + 4 * slot);    // LHH
    cswa(bands[4], out, 2, out + 5 * slot);    // HLH
    cswa(bands[5], out, 2, out + 6 * slot);    // HHL
    cswa(bands[6], out, 3, out + 7 * slot);    // HHH
}
